// Attention_1967095021915
// MI455X (gfx1250) — matrix operand order repaired and checked by reading, not compiled, not run on hardware
//
#include <hip/hip_runtime.h>

typedef __attribute__((ext_vector_type(16))) __bf16 v16bf;
typedef __attribute__((ext_vector_type(8)))  float  v8f;

#define HEADS 16
#define DHEAD 64
#define DIM   1024
#define INNER 1024
#define NSEQ  2048
#define BATCH 4
#define NKV   2049   // null kv + 2048
#define NKVP  2080   // padded to multiple of 32
#define NROWS (BATCH * NSEQ)

// ---------- CDNA5 async memory->LDS (ASYNCcnt-tracked), via inline asm ----------
__device__ __forceinline__ void async_ld_b128(unsigned ldsOff, const void* g) {
    asm volatile("global_load_async_to_lds_b128 %0, %1, off"
                 :: "v"(ldsOff), "v"((unsigned long long)(size_t)g)
                 : "memory");
}
#define S_WAIT_ASYNC(n) asm volatile("s_wait_asynccnt " #n ::: "memory")

__device__ __forceinline__ unsigned lds_off(const void* p) {
    return (unsigned)(size_t)p;   // LDS aperture: low 32 bits = LDS byte address
}

// ---------- helpers ----------
__device__ __forceinline__ __bf16 f2bf(float f) {
    unsigned u = __builtin_bit_cast(unsigned, f);
    u += 0x7FFFu + ((u >> 16) & 1u);              // round-to-nearest-even
    unsigned short h = (unsigned short)(u >> 16);
    return __builtin_bit_cast(__bf16, h);
}

__device__ __forceinline__ float wave_red_sum32(float v) {
#pragma unroll
    for (int m = 1; m < 32; m <<= 1) v += __shfl_xor(v, m, 32);
    return v;
}

union U16 { uint4 u[2]; v16bf v; };

// ---------- weight transpose + f32->bf16 : in[K][N] -> out[N][K] ----------
__global__ void __launch_bounds__(256)
transpose_cvt_kernel(const float* __restrict__ in, __bf16* __restrict__ outT,
                     int K, int N) {
    __shared__ float sm[32][33];
    int n0 = blockIdx.x * 32, k0 = blockIdx.y * 32;
    int t = threadIdx.x;
    {
        int r = t >> 3, c = (t & 7) * 4;
        float4 v = *(const float4*)(in + (size_t)(k0 + r) * N + n0 + c);
        sm[r][c] = v.x; sm[r][c + 1] = v.y; sm[r][c + 2] = v.z; sm[r][c + 3] = v.w;
    }
    __syncthreads();
    {
        int rn = t >> 3, ks = (t & 7) * 4;
        union { uint2 u; __bf16 e[4]; } pk;
#pragma unroll
        for (int i = 0; i < 4; i++) pk.e[i] = f2bf(sm[ks + i][rn]);
        *(uint2*)(outT + (size_t)(n0 + rn) * K + k0 + ks) = pk.u;
    }
}

// ---------- LayerNorm (1024 wide) -> bf16 ----------
__global__ void __launch_bounds__(256)
layernorm_bf16_kernel(const float* __restrict__ x, const float* __restrict__ gamma,
                      __bf16* __restrict__ xn) {
    __shared__ float red[16];
    int row = blockIdx.x, t = threadIdx.x;
    int wave = t >> 5, lane = t & 31;
    const float4* xr = (const float4*)(x + (size_t)row * DIM);
    float4 v = xr[t];
    float s  = v.x + v.y + v.z + v.w;
    float ss = v.x * v.x + v.y * v.y + v.z * v.z + v.w * v.w;
    s = wave_red_sum32(s);
    ss = wave_red_sum32(ss);
    if (lane == 0) { red[wave] = s; red[8 + wave] = ss; }
    __syncthreads();
    if (t == 0) {
        float a = 0.f, b = 0.f;
#pragma unroll
        for (int i = 0; i < 8; i++) { a += red[i]; b += red[8 + i]; }
        red[0] = a; red[8] = b;
    }
    __syncthreads();
    float mu   = red[0] * (1.f / DIM);
    float var  = red[8] * (1.f / DIM) - mu * mu;
    float rstd = rsqrtf(var + 1e-5f);
    float4 g = ((const float4*)gamma)[t];
    __bf16* o = xn + (size_t)row * DIM + t * 4;
    o[0] = f2bf((v.x - mu) * rstd * g.x);
    o[1] = f2bf((v.y - mu) * rstd * g.y);
    o[2] = f2bf((v.z - mu) * rstd * g.z);
    o[3] = f2bf((v.w - mu) * rstd * g.w);
}

// ---------- bf16 WMMA GEMM: C[M,N] = A[M,K] * Bt[N,K]^T, f32 out ----------
// 256 threads (8 waves); tile 256(M) x 64(N); K-step 32; async double-buffered LDS.
#define GM_BM 256
#define GM_BN 64
#define GM_BK 32

__global__ void __launch_bounds__(256)
gemm_bf16_kernel(const __bf16* __restrict__ A, const __bf16* __restrict__ Bt,
                 float* __restrict__ C, int M, int N, int K) {
    __shared__ __align__(16) __bf16 smA[2][GM_BM * GM_BK];  // 2 x 16 KB
    __shared__ __align__(16) __bf16 smB[2][GM_BN * GM_BK];  // 2 x 4 KB, [n][k]
    int t = threadIdx.x, wave = t >> 5, lane = t & 31;
    int row16 = lane & 15, hi = lane >> 4;
    int kbase = hi * 8;
    int m0 = blockIdx.y * GM_BM, n0 = blockIdx.x * GM_BN;

    v8f acc[2][4];
#pragma unroll
    for (int mm = 0; mm < 2; mm++)
#pragma unroll
        for (int nn = 0; nn < 4; nn++)
#pragma unroll
            for (int r = 0; r < 8; r++) acc[mm][nn][r] = 0.f;

    // per-thread staging coordinates
    int ar[4], as[4];
#pragma unroll
    for (int i = 0; i < 4; i++) { int idx = t + i * 256; ar[i] = idx >> 2; as[i] = (idx & 3) * 8; }
    int br = t >> 2, bs = (t & 3) * 8;

    const int T = K / GM_BK;

    auto issue = [&](int k0, int buf) {
#pragma unroll
        for (int i = 0; i < 4; i++)
            async_ld_b128(lds_off(&smA[buf][ar[i] * GM_BK + as[i]]),
                          A + (size_t)(m0 + ar[i]) * K + k0 + as[i]);
        async_ld_b128(lds_off(&smB[buf][br * GM_BK + bs]),
                      Bt + (size_t)(n0 + br) * K + k0 + bs);
    };

    auto compute = [&](int it) {
        const __bf16* bufA = smA[it & 1];
        const __bf16* bufB = smB[it & 1];
        U16 af[2];
#pragma unroll
        for (int mm = 0; mm < 2; mm++) {
            const uint4* pa = (const uint4*)(bufA + (wave * 32 + mm * 16 + row16) * GM_BK + kbase);
            af[mm].u[0] = pa[0];   // K = kbase..kbase+7
            af[mm].u[1] = pa[2];   // K = kbase+16..kbase+23
        }
#pragma unroll
        for (int nn = 0; nn < 4; nn++) {
            U16 bf;
            const uint4* pb = (const uint4*)(bufB + (nn * 16 + row16) * GM_BK + hi * 8);
            bf.u[0] = pb[0];
            bf.u[1] = pb[2];
#pragma unroll
            for (int mm = 0; mm < 2; mm++)
                acc[mm][nn] = __builtin_amdgcn_wmma_f32_16x16x32_bf16(
                    false, af[mm].v, false, bf.v, (short)0, acc[mm][nn], false, false);
        }
    };

    issue(0, 0);
    for (int it = 0; it < T - 1; it++) {       // steady state: branch-free
        __syncthreads();                       // readers of buf[(it+1)&1] done
        issue((it + 1) * GM_BK, (it + 1) & 1);
        S_WAIT_ASYNC(5);                       // tile `it` landed; next in flight
        __syncthreads();                       // tile `it` visible to all waves
        compute(it);
    }
    __syncthreads();                           // drain: last tile
    S_WAIT_ASYNC(0);
    __syncthreads();
    compute(T - 1);

#pragma unroll
    for (int mm = 0; mm < 2; mm++)
#pragma unroll
        for (int nn = 0; nn < 4; nn++)
#pragma unroll
            for (int r = 0; r < 8; r++) {
                int row = m0 + wave * 32 + mm * 16 + hi * 8 + r;
                int col = n0 + nn * 16 + row16;
                C[(size_t)row * N + col] = acc[mm][nn][r];
            }
}

// ---------- per-(row,head) l2norm of q,k : 1 wave per block ----------
__global__ void qkv_norm_kernel(const float* __restrict__ q_raw,
                                const float* __restrict__ kv_raw,
                                const float* __restrict__ q_scale,
                                const float* __restrict__ k_scale,
                                __bf16* __restrict__ qn, __bf16* __restrict__ kn) {
    int blk = blockIdx.x;
    int h = blk & 15;
    size_t row = (size_t)(blk >> 4);
    int b = (int)(row >> 11), rn = (int)(row & 2047);
    int t = threadIdx.x;

    const float* qp = q_raw + row * INNER + h * DHEAD;
    float q0 = qp[t], q1 = qp[t + 32];
    float qs = wave_red_sum32(q0 * q0 + q1 * q1);
    float qi = 1.f / fmaxf(sqrtf(qs), 1e-12f);
    __bf16* qo = qn + row * INNER + h * DHEAD;
    qo[t]      = f2bf(q0 * qi * q_scale[t]);
    qo[t + 32] = f2bf(q1 * qi * q_scale[t + 32]);

    const float* kp = kv_raw + row * (2 * INNER) + h * DHEAD;
    float k0 = kp[t], k1 = kp[t + 32];
    float ks = wave_red_sum32(k0 * k0 + k1 * k1);
    float ki = 1.f / fmaxf(sqrtf(ks), 1e-12f);
    size_t kidx = (((size_t)(b * HEADS + h)) * NKV + 1 + rn) * DHEAD;
    kn[kidx + t]      = f2bf(k0 * ki * k_scale[t]);
    kn[kidx + t + 32] = f2bf(k1 * ki * k_scale[t + 32]);
}

// ---------- V transpose: kv_raw v-part [row][d] -> vt [b,h][d][key(2080 pad)] ----------
__global__ void __launch_bounds__(256)
v_transpose_kernel(const float* __restrict__ kv_raw, __bf16* __restrict__ vt) {
    __shared__ float sm[32][65];                 // [key][d]
    int blk = blockIdx.x;
    int tile = blk % (NKVP / 32);
    int bh = blk / (NKVP / 32);
    int b = bh >> 4, h = bh & 15;
    int j0 = tile * 32;
    int t = threadIdx.x;
    {
        int key = t >> 3, ds = (t & 7) * 8;
        int j = j0 + key;
        if (j >= 1 && j <= NSEQ) {
            const float* src = kv_raw + ((size_t)(b * NSEQ + j - 1)) * (2 * INNER)
                               + INNER + h * DHEAD + ds;
#pragma unroll
            for (int i = 0; i < 8; i++) sm[key][ds + i] = src[i];
        } else {
#pragma unroll
            for (int i = 0; i < 8; i++) sm[key][ds + i] = 0.f;
        }
    }
    __syncthreads();
    {
        int d = t >> 2, c = (t & 3) * 8;
        union { uint4 u; __bf16 e[8]; } pk;
#pragma unroll
        for (int i = 0; i < 8; i++) pk.e[i] = f2bf(sm[c + i][d]);
        *(uint4*)(vt + ((size_t)bh * DHEAD + d) * NKVP + j0 + c) = pk.u;
    }
}

// ---------- null kv slot (key = 0) ----------
__global__ void nullkv_kernel(const float* __restrict__ null_kv,
                              const float* __restrict__ k_scale,
                              __bf16* __restrict__ kn, __bf16* __restrict__ vt) {
    int bh = blockIdx.x;
    int h = bh & 15, t = threadIdx.x;
    const float* nk = null_kv + h * DHEAD;
    const float* nv = null_kv + HEADS * DHEAD + h * DHEAD;
    float k0 = nk[t], k1 = nk[t + 32];
    float ss = wave_red_sum32(k0 * k0 + k1 * k1);
    float inv = 1.f / fmaxf(sqrtf(ss), 1e-12f);
    size_t idx = (size_t)bh * NKV * DHEAD;
    kn[idx + t]      = f2bf(k0 * inv * k_scale[t]);
    kn[idx + t + 32] = f2bf(k1 * inv * k_scale[t + 32]);
    vt[((size_t)bh * DHEAD + t) * NKVP]      = f2bf(nv[t]);
    vt[((size_t)bh * DHEAD + t + 32) * NKVP] = f2bf(nv[t + 32]);
}

// ---------- flash attention per (b,h, 128-row q block) ----------
__global__ void __launch_bounds__(256)
flash_attn_kernel(const __bf16* __restrict__ qn, const __bf16* __restrict__ kn,
                  const __bf16* __restrict__ vt, const unsigned char* __restrict__ cmask,
                  __bf16* __restrict__ attn_out) {
    __shared__ __align__(16) __bf16 smK[2][32 * 64];   // [key][d]   2 x 4 KB
    __shared__ __align__(16) __bf16 smV[2][64 * 32];   // [d][key]   2 x 4 KB
    __shared__ __align__(16) __bf16 smP[8 * 16 * 32];  // per-wave P bounce, 8 KB

    int bh = blockIdx.x >> 4;
    int qb = blockIdx.x & 15;
    int b = bh >> 4, h = bh & 15;
    int t = threadIdx.x, wave = t >> 5, lane = t & 31;
    int row16 = lane & 15, hi = lane >> 4;
    int kbase = hi * 8;

    // Q fragments (16 rows x 64 head-dim = two 16x32 A-frags)
    int qrow_lane = qb * 128 + wave * 16 + row16;
    const __bf16* qp = qn + ((size_t)(b * NSEQ + qrow_lane)) * INNER + h * DHEAD;
    U16 qa0, qa1;
    qa0.u[0] = *(const uint4*)(qp + kbase);
    qa0.u[1] = *(const uint4*)(qp + kbase + 16);
    qa1.u[0] = *(const uint4*)(qp + 32 + kbase);
    qa1.u[1] = *(const uint4*)(qp + 32 + kbase + 16);

    v8f acc[4];
#pragma unroll
    for (int nn = 0; nn < 4; nn++)
#pragma unroll
        for (int r = 0; r < 8; r++) acc[nn][r] = 0.f;
    float mI[8], lI[8];
#pragma unroll
    for (int r = 0; r < 8; r++) { mI[r] = -3.0e38f; lI[r] = 0.f; }

    const __bf16* kbp = kn + (size_t)bh * NKV * DHEAD;
    const __bf16* vbp = vt + (size_t)bh * DHEAD * NKVP;
    __bf16* myP = smP + wave * 16 * 32;

    // per-thread staging coordinates
    int kr = t >> 3, ks2 = (t & 7) * 8;     // K tile: [key][d]
    int vd = t >> 2, vs = (t & 3) * 8;      // V tile: [d][key]

    auto issueKV = [&](int j0, int buf) {
        int j = j0 + kr; if (j > NKV - 1) j = NKV - 1;   // clamp; garbage keys masked later
        async_ld_b128(lds_off(&smK[buf][kr * 64 + ks2]), kbp + (size_t)j * DHEAD + ks2);
        async_ld_b128(lds_off(&smV[buf][vd * 32 + vs]),  vbp + (size_t)vd * NKVP + j0 + vs);
    };

    auto compute = [&](int it) {
        int j0 = it * 32;
        const __bf16* bK = smK[it & 1];
        const __bf16* bV = smV[it & 1];

        // S = Q * K^T for two 16-col sub-tiles
        v8f sfr[2];
        bool okx[2];
#pragma unroll
        for (int x = 0; x < 2; x++) {
            const __bf16* pk0 = bK + (x * 16 + row16) * 64 + hi * 8;   // d 0..31
            const __bf16* pk1 = pk0 + 32;                              // d 32..63
            U16 b0, b1;
            b0.u[0] = ((const uint4*)pk0)[0]; b0.u[1] = ((const uint4*)pk0)[2];
            b1.u[0] = ((const uint4*)pk1)[0]; b1.u[1] = ((const uint4*)pk1)[2];
            v8f sz;
#pragma unroll
            for (int r = 0; r < 8; r++) sz[r] = 0.f;
            sz = __builtin_amdgcn_wmma_f32_16x16x32_bf16(false, qa0.v, false, b0.v,
                                                         (short)0, sz, false, false);
            sz = __builtin_amdgcn_wmma_f32_16x16x32_bf16(false, qa1.v, false, b1.v,
                                                         (short)0, sz, false, false);
            sfr[x] = sz;
            int cg = j0 + x * 16 + row16;
            okx[x] = (cg < NKV) && (cg == 0 || cmask[(size_t)b * NSEQ + cg - 1] != 0);
        }

        // online softmax (per C-row r; 16-lane group reductions)
        float pscale[8], p0a[8], p1a[8];
#pragma unroll
        for (int r = 0; r < 8; r++) {
            float s0 = okx[0] ? sfr[0][r] * 8.0f : -1.0e30f;
            float s1 = okx[1] ? sfr[1][r] * 8.0f : -1.0e30f;
            float rm = fmaxf(s0, s1);
#pragma unroll
            for (int m = 1; m < 16; m <<= 1) rm = fmaxf(rm, __shfl_xor(rm, m, 32));
            float mnew = fmaxf(mI[r], rm);
            float sc = __expf(mI[r] - mnew);
            float e0 = __expf(s0 - mnew);
            float e1 = __expf(s1 - mnew);
            float rs = e0 + e1;
#pragma unroll
            for (int m = 1; m < 16; m <<= 1) rs += __shfl_xor(rs, m, 32);
            lI[r] = lI[r] * sc + rs;
            mI[r] = mnew;
            pscale[r] = sc;
            p0a[r] = e0; p1a[r] = e1;
        }
#pragma unroll
        for (int nn = 0; nn < 4; nn++)
#pragma unroll
            for (int r = 0; r < 8; r++) acc[nn][r] *= pscale[r];

        // P: C-layout -> LDS -> A-layout (per-wave private region, DS in-order)
#pragma unroll
        for (int r = 0; r < 8; r++) {
            int prow = hi * 8 + r;
            myP[prow * 32 + row16]      = f2bf(p0a[r]);
            myP[prow * 32 + 16 + row16] = f2bf(p1a[r]);
        }
        U16 pa;
        const uint4* pp = (const uint4*)(myP + row16 * 32 + kbase);
        pa.u[0] = pp[0];
        pa.u[1] = pp[2];

        // acc += P(16x32) * V(32x64)
#pragma unroll
        for (int nn = 0; nn < 4; nn++) {
            U16 vb;
            const uint4* pv = (const uint4*)(bV + (nn * 16 + row16) * 32 + hi * 8);
            vb.u[0] = pv[0];
            vb.u[1] = pv[2];
            acc[nn] = __builtin_amdgcn_wmma_f32_16x16x32_bf16(
                false, pa.v, false, vb.v, (short)0, acc[nn], false, false);
        }
    };

    const int T = NKVP / 32;   // 65 tiles
    issueKV(0, 0);
    for (int it = 0; it < T - 1; it++) {       // steady state: branch-free
        __syncthreads();
        issueKV((it + 1) * 32, (it + 1) & 1);
        S_WAIT_ASYNC(2);
        __syncthreads();
        compute(it);
    }
    __syncthreads();                           // drain: last tile
    S_WAIT_ASYNC(0);
    __syncthreads();
    compute(T - 1);

    // epilogue: divide by l, write bf16 [row, h*64 + d]
#pragma unroll
    for (int nn = 0; nn < 4; nn++)
#pragma unroll
        for (int r = 0; r < 8; r++) {
            int orow = qb * 128 + wave * 16 + hi * 8 + r;
            float o = acc[nn][r] / lI[r];
            attn_out[((size_t)(b * NSEQ + orow)) * INNER + h * DHEAD + nn * 16 + row16] =
                f2bf(o);
        }
}

// ---------- host launcher ----------
extern "C" void kernel_launch(void* const* d_in, const int* in_sizes, int n_in,
                              void* d_out, int out_size, void* d_ws, size_t ws_size,
                              hipStream_t stream) {
    (void)in_sizes; (void)n_in; (void)out_size; (void)ws_size;
    const float* x        = (const float*)d_in[0];
    const unsigned char* cmask = (const unsigned char*)d_in[1];
    const float* gamma    = (const float*)d_in[2];
    const float* wq       = (const float*)d_in[3];
    const float* wkv      = (const float*)d_in[4];
    const float* null_kv  = (const float*)d_in[5];
    const float* q_scale  = (const float*)d_in[6];
    const float* k_scale  = (const float*)d_in[7];
    const float* wo       = (const float*)d_in[8];
    float* out = (float*)d_out;

    char* ws = (char*)d_ws;
    size_t off = 0;
    auto carve = [&](size_t bytes) { char* p = ws + off; off += (bytes + 255) & ~(size_t)255; return p; };
    __bf16* xn_b   = (__bf16*)carve((size_t)NROWS * DIM * 2);
    __bf16* wqT    = (__bf16*)carve((size_t)INNER * DIM * 2);
    __bf16* wkvT   = (__bf16*)carve((size_t)2 * INNER * DIM * 2);
    __bf16* woT    = (__bf16*)carve((size_t)DIM * INNER * 2);
    float*  q_raw  = (float*)carve((size_t)NROWS * INNER * 4);
    float*  kv_raw = (float*)carve((size_t)NROWS * 2 * INNER * 4);
    __bf16* qn     = (__bf16*)carve((size_t)NROWS * INNER * 2);
    __bf16* kn     = (__bf16*)carve((size_t)BATCH * HEADS * NKV * DHEAD * 2);
    __bf16* vt     = (__bf16*)carve((size_t)BATCH * HEADS * DHEAD * NKVP * 2);
    __bf16* attn_b = (__bf16*)carve((size_t)NROWS * INNER * 2);

    transpose_cvt_kernel<<<dim3(INNER / 32, DIM / 32), 256, 0, stream>>>(wq, wqT, DIM, INNER);
    transpose_cvt_kernel<<<dim3(2 * INNER / 32, DIM / 32), 256, 0, stream>>>(wkv, wkvT, DIM, 2 * INNER);
    transpose_cvt_kernel<<<dim3(DIM / 32, INNER / 32), 256, 0, stream>>>(wo, woT, INNER, DIM);

    layernorm_bf16_kernel<<<NROWS, 256, 0, stream>>>(x, gamma, xn_b);

    gemm_bf16_kernel<<<dim3(INNER / GM_BN, NROWS / GM_BM), 256, 0, stream>>>(
        xn_b, wqT, q_raw, NROWS, INNER, DIM);
    gemm_bf16_kernel<<<dim3(2 * INNER / GM_BN, NROWS / GM_BM), 256, 0, stream>>>(
        xn_b, wkvT, kv_raw, NROWS, 2 * INNER, DIM);

    v_transpose_kernel<<<BATCH * HEADS * (NKVP / 32), 256, 0, stream>>>(kv_raw, vt);
    nullkv_kernel<<<BATCH * HEADS, 32, 0, stream>>>(null_kv, k_scale, kn, vt);
    qkv_norm_kernel<<<NROWS * HEADS, 32, 0, stream>>>(q_raw, kv_raw, q_scale, k_scale, qn, kn);

    flash_attn_kernel<<<BATCH * HEADS * (NSEQ / 128), 256, 0, stream>>>(
        qn, kn, vt, cmask, attn_b);

    gemm_bf16_kernel<<<dim3(DIM / GM_BN, NROWS / GM_BM), 256, 0, stream>>>(
        attn_b, woT, out, NROWS, DIM, DIM);
}
